// GCN_RW_full_13975823581634
// MI455X (gfx1250) — compile-verified, matrix operand order checked by reading, not run on hardware
//
#include <hip/hip_runtime.h>
#include <hip/hip_bf16.h>

// ---------------------------------------------------------------------------
// Problem constants (match the JAX reference)
// ---------------------------------------------------------------------------
constexpr int NN   = 100000;  // nodes
constexpr int FEAT = 256;     // input features
constexpr int HID  = 128;     // hidden
constexpr int CLS  = 40;      // classes
constexpr int KHOP = 4;       // hops
constexpr int RWS  = 10;      // random-walk samples per node
constexpr int NL   = 2;       // layers

// ---------------------------------------------------------------------------
// bf16 helpers (raw-bit ushort storage; round-to-nearest-even convert)
// ---------------------------------------------------------------------------
__device__ __forceinline__ unsigned short f2bf(float f) {
    union { float f; unsigned int u; } v; v.f = f;
    unsigned int r = v.u + 0x7FFFu + ((v.u >> 16) & 1u);
    return (unsigned short)(r >> 16);
}
__device__ __forceinline__ float bf_lo(unsigned int u) {
    union { unsigned int u; float f; } v; v.u = u << 16; return v.f;
}
__device__ __forceinline__ float bf_hi(unsigned int u) {
    union { unsigned int u; float f; } v; v.u = u & 0xFFFF0000u; return v.f;
}

// ---------------------------------------------------------------------------
// WMMA types (CDNA5 wave32)
// ---------------------------------------------------------------------------
typedef __attribute__((ext_vector_type(16))) __bf16 v16bf;
typedef __attribute__((ext_vector_type(8)))  float  v8f;

union Frag16 {                 // 16 bf16 elements = 8 VGPRs; dword v holds pair (k0,k0+1)
    unsigned int u[8];
    v16bf        v;
};

// LDS offset of a __shared__ object: flat shared addresses carry the LDS byte
// offset in their low 32 bits.
__device__ __forceinline__ unsigned int lds_offset(const void* p) {
    return (unsigned int)(unsigned long long)p;
}

// ---------------------------------------------------------------------------
// GEMM:  Out[M,Nn] = A[M,Kd] @ W[Kd,Nn] + bias[Nn]
//   A_BF16:  A is bf16 row-major -> staged with global_load_async_to_lds_b128
//   !A_BF16: A is fp32 row-major -> VALU f32->bf16 convert on the LDS fill
//   OUT_BF16: store bf16 (ushort) else fp32
// Block tile 64x64, 8 waves; wave w -> rows [(w&3)*16,+16), cols [(w>>2)*32,+32)
// (two 16x16 WMMA tiles per wave sharing the A fragment).
// ---------------------------------------------------------------------------
template <bool A_BF16, bool OUT_BF16>
__global__ __launch_bounds__(256)
void gemm_bias_wmma(const void* __restrict__ Aptr,
                    const float* __restrict__ W,
                    const float* __restrict__ bias,
                    void* __restrict__ Out,
                    int M, int Kd, int Nn)
{
    // A stride: 40 halfwords = 80 B/row. Row starts 16B-aligned (async B128
    // LDS writes), and 20 dwords/row -> 20*r mod 64 covers 16 distinct banks
    // for r=0..15: conflict-free fragment reads.
    constexpr int AST = 40;
    constexpr int BST = 34;   // 17-bank stride for B fragment reads
    __shared__ unsigned short As[64 * AST];   // [row][k]   64x32 bf16 tile
    __shared__ unsigned short Bs[64 * BST];   // [col][k]   (W transposed) 64x32

    const int t    = threadIdx.x;
    const int wave = t >> 5;
    const int lane = t & 31;
    const int mBase = blockIdx.x * 64;
    const int nBase = blockIdx.y * 64;

    const int half = lane >> 4;       // K-half selector per ISA 16-bit layout
    const int r16  = lane & 15;       // row (A) / col (B) within 16
    const int mrow = (wave & 3) * 16;
    const int ncol = (wave >> 2) * 32;

    v8f acc0 = {0.f, 0.f, 0.f, 0.f, 0.f, 0.f, 0.f, 0.f};
    v8f acc1 = {0.f, 0.f, 0.f, 0.f, 0.f, 0.f, 0.f, 0.f};

    // Per-thread A-tile staging coordinates: 256 threads x 8 bf16 (16 B) each.
    const int rl = t >> 2;           // 0..63  (tile row)
    const int kk = (t & 3) * 8;      // 0,8,16,24 (k offset)
    int arow = mBase + rl; if (arow >= M) arow = M - 1;   // clamp (stores masked)

    for (int kB = 0; kB < Kd; kB += 32) {
        // ---- stage A tile: 64 rows x 32 k ----
        if constexpr (A_BF16) {
            // Direct global -> LDS async copy (ASYNCcnt-tracked), 16 B / lane.
            const unsigned short* gsrc =
                (const unsigned short*)Aptr + (size_t)arow * Kd + kB + kk;
            const unsigned long long gaddr = (unsigned long long)gsrc;
            const unsigned int dstoff = lds_offset(&As[rl * AST + kk]);
            asm volatile("global_load_async_to_lds_b128 %0, %1, off"
                         :: "v"(dstoff), "v"(gaddr) : "memory");
        } else {
            const float* Af = (const float*)Aptr;
            const float4 p0 = *(const float4*)(Af + (size_t)arow * Kd + kB + kk);
            const float4 p1 = *(const float4*)(Af + (size_t)arow * Kd + kB + kk + 4);
            unsigned short* d = &As[rl * AST + kk];
            d[0] = f2bf(p0.x); d[1] = f2bf(p0.y); d[2] = f2bf(p0.z); d[3] = f2bf(p0.w);
            d[4] = f2bf(p1.x); d[5] = f2bf(p1.y); d[6] = f2bf(p1.z); d[7] = f2bf(p1.w);
        }
        // ---- stage W tile transposed: Bs[col][k], zero-pad cols >= Nn ----
        {
            const int c   = t >> 2;          // 0..63
            const int kw  = (t & 3) * 8;
            const int col = nBase + c;
            unsigned short* d = &Bs[c * BST + kw];
            #pragma unroll
            for (int j = 0; j < 8; ++j) {
                const int k = kB + kw + j;
                float v = (col < Nn) ? W[(size_t)k * Nn + col] : 0.f;
                d[j] = f2bf(v);
            }
        }
        if constexpr (A_BF16) {
            asm volatile("s_wait_asynccnt 0x0" ::: "memory");
        }
        __syncthreads();

        // ---- build fragments per ISA 16-bit A/B VGPR layout and run WMMA ----
        Frag16 a, b0, b1;
        const unsigned short* ap  = &As[(mrow + r16) * AST];
        const unsigned short* bp0 = &Bs[(ncol + r16) * BST];
        const unsigned short* bp1 = &Bs[(ncol + 16 + r16) * BST];
        #pragma unroll
        for (int v = 0; v < 8; ++v) {
            const int k0 = ((v >> 2) << 4) + (half << 3) + ((v & 3) << 1);
            a.u[v]  = *(const unsigned int*)(ap  + k0);
            b0.u[v] = *(const unsigned int*)(bp0 + k0);
            b1.u[v] = *(const unsigned int*)(bp1 + k0);
        }
        acc0 = __builtin_amdgcn_wmma_f32_16x16x32_bf16(false, a.v, false, b0.v,
                                                       (short)0, acc0, false, false);
        acc1 = __builtin_amdgcn_wmma_f32_16x16x32_bf16(false, a.v, false, b1.v,
                                                       (short)0, acc1, false, false);
        __syncthreads();
    }

    // ---- epilogue: bias add, masked store (C layout: row = v + 8*half, col = r16) ----
    #pragma unroll
    for (int tile = 0; tile < 2; ++tile) {
        const v8f& acc = tile ? acc1 : acc0;
        const int col = nBase + ncol + tile * 16 + r16;
        if (col < Nn) {
            const float bv = bias[col];
            #pragma unroll
            for (int v = 0; v < 8; ++v) {
                const int row = mBase + mrow + half * 8 + v;
                if (row < M) {
                    const float val = acc[v] + bv;
                    if constexpr (OUT_BF16)
                        ((unsigned short*)Out)[(size_t)row * Nn + col] = f2bf(val);
                    else
                        ((float*)Out)[(size_t)row * Nn + col] = val;
                }
            }
        }
    }
}

// ---------------------------------------------------------------------------
// Random-walk gather + attention aggregate + ReLU.
// One wave per node; each lane owns 4 features (8-byte bf16 load per row).
// The 40 (endpoint, weight) pairs per node are loaded/computed ONCE by the
// wave (lanes cover flat indices {lane, lane+32}) and broadcast via __shfl,
// instead of 32x-redundant per-lane loads + rsqrtf.
//   agg = att[0]*h[n] + sum_k att[k] * mean_s( sqrt(d[n])/sqrt(d[e]) * h[e] )
// ---------------------------------------------------------------------------
__global__ __launch_bounds__(256)
void gather_agg_relu(const unsigned short* __restrict__ hlin,
                     const float* __restrict__ degree,
                     const int*  __restrict__ ends_l,   // [KHOP][NN*RWS] for this layer
                     const float* __restrict__ att_l,   // [KHOP+1] for this layer
                     unsigned short* __restrict__ hout)
{
    const int wid  = (int)((blockIdx.x * blockDim.x + threadIdx.x) >> 5);
    const int lane = threadIdx.x & 31;
    if (wid >= NN) return;
    const int n = wid;
    const int f = lane * 4;

    const float dsn = sqrtf(degree[n]);

    // Preload all KHOP*RWS = 40 endpoint indices + degree weights (2 per lane).
    int eA = 0; float wA = 0.f;
    {
        const int idx = lane;                       // 0..31 -> all valid (<40)
        const int k = idx / RWS, s = idx - k * RWS;
        eA = ends_l[(size_t)k * ((size_t)NN * RWS) + (size_t)n * RWS + s];
        wA = dsn * rsqrtf(degree[eA]);
    }
    int eB = 0; float wB = 0.f;
    {
        const int idx = lane + 32;                  // 32..63 -> valid if < 40
        if (idx < KHOP * RWS) {
            const int k = idx / RWS, s = idx - k * RWS;
            eB = ends_l[(size_t)k * ((size_t)NN * RWS) + (size_t)n * RWS + s];
            wB = dsn * rsqrtf(degree[eB]);
        }
    }

    const uint2 p = *(const uint2*)(hlin + (size_t)n * HID + f);
    const float a0 = att_l[0];
    float acc0 = a0 * bf_lo(p.x), acc1 = a0 * bf_hi(p.x);
    float acc2 = a0 * bf_lo(p.y), acc3 = a0 * bf_hi(p.y);

    const float inv = 1.0f / (float)RWS;

    #pragma unroll
    for (int k = 0; k < KHOP; ++k) {
        float s0 = 0.f, s1 = 0.f, s2 = 0.f, s3 = 0.f;
        #pragma unroll
        for (int s = 0; s < RWS; ++s) {
            const int flat = k * RWS + s;           // compile-time constant
            const int   e = (flat < 32) ? __shfl(eA, flat, 32)
                                        : __shfl(eB, flat - 32, 32);
            const float w = (flat < 32) ? __shfl(wA, flat, 32)
                                        : __shfl(wB, flat - 32, 32);
            const uint2 q = *(const uint2*)(hlin + (size_t)e * HID + f);
            s0 += w * bf_lo(q.x);  s1 += w * bf_hi(q.x);
            s2 += w * bf_lo(q.y);  s3 += w * bf_hi(q.y);
        }
        const float ak = att_l[k + 1] * inv;
        acc0 += ak * s0;  acc1 += ak * s1;
        acc2 += ak * s2;  acc3 += ak * s3;
    }

    acc0 = fmaxf(acc0, 0.f);  acc1 = fmaxf(acc1, 0.f);
    acc2 = fmaxf(acc2, 0.f);  acc3 = fmaxf(acc3, 0.f);
    uint2 o;
    o.x = (unsigned int)f2bf(acc0) | ((unsigned int)f2bf(acc1) << 16);
    o.y = (unsigned int)f2bf(acc2) | ((unsigned int)f2bf(acc3) << 16);
    *(uint2*)(hout + (size_t)n * HID + f) = o;
}

// ---------------------------------------------------------------------------
// In-place log_softmax over CLS=40 columns; one wave per row,
// lane covers cols {lane, lane+32}; wave32 shuffle reductions.
// ---------------------------------------------------------------------------
__global__ __launch_bounds__(256)
void log_softmax_inplace(float* __restrict__ logits)
{
    const int wid  = (int)((blockIdx.x * blockDim.x + threadIdx.x) >> 5);
    const int lane = threadIdx.x & 31;
    if (wid >= NN) return;
    float* rp = logits + (size_t)wid * CLS;

    const bool hi = (lane + 32) < CLS;
    const float x0 = rp[lane];
    const float x1 = hi ? rp[lane + 32] : -3.402823466e38f;

    float m = fmaxf(x0, x1);
    #pragma unroll
    for (int off = 16; off > 0; off >>= 1) m = fmaxf(m, __shfl_xor(m, off, 32));

    float s = __expf(x0 - m) + (hi ? __expf(x1 - m) : 0.f);
    #pragma unroll
    for (int off = 16; off > 0; off >>= 1) s += __shfl_xor(s, off, 32);

    const float lse = m + __logf(s);
    rp[lane] = x0 - lse;
    if (hi) rp[lane + 32] = x1 - lse;
}

// ---------------------------------------------------------------------------
// Launch: GEMM1 -> gather L0 -> GEMM2 -> gather L1 -> GEMM3 -> log_softmax
// Workspace: two bf16 [NN,HID] ping-pong buffers (25.6 MB each).
// Logits go straight to d_out (fp32 [NN,CLS]) and are softmaxed in place.
// ---------------------------------------------------------------------------
extern "C" void kernel_launch(void* const* d_in, const int* in_sizes, int n_in,
                              void* d_out, int out_size, void* d_ws, size_t ws_size,
                              hipStream_t stream)
{
    const float* x      = (const float*)d_in[0];
    const float* degree = (const float*)d_in[1];
    const int*   ends   = (const int*)  d_in[2];   // [NL, KHOP, NN*RWS]
    const float* att    = (const float*)d_in[3];   // [NL, KHOP+1]
    const float* W0     = (const float*)d_in[4];
    const float* b0     = (const float*)d_in[5];
    const float* W1     = (const float*)d_in[6];
    const float* b1     = (const float*)d_in[7];
    const float* W2     = (const float*)d_in[8];
    const float* b2     = (const float*)d_in[9];

    unsigned short* ws0 = (unsigned short*)d_ws;            // bf16 [NN, HID]
    unsigned short* ws1 = ws0 + (size_t)NN * HID;           // bf16 [NN, HID]
    float* logits = (float*)d_out;                          // fp32 [NN, CLS]

    const dim3 blk(256);
    const dim3 gGemmH((NN + 63) / 64, HID / 64);            // N = 128 -> y = 2
    const dim3 gGemmC((NN + 63) / 64, 1);                   // N = 40  -> one 64-col tile
    const dim3 gWave((NN + 7) / 8);                         // 8 waves / block, 1 node / wave

    const size_t endsLayerStride = (size_t)KHOP * NN * RWS; // 4,000,000

    // Layer 0
    gemm_bias_wmma<false, true><<<gGemmH, blk, 0, stream>>>(x, W0, b0, ws0, NN, FEAT, HID);
    gather_agg_relu<<<gWave, blk, 0, stream>>>(ws0, degree, ends, att, ws1);

    // Layer 1
    gemm_bias_wmma<true, true><<<gGemmH, blk, 0, stream>>>(ws1, W1, b1, ws0, NN, HID, HID);
    gather_agg_relu<<<gWave, blk, 0, stream>>>(ws0, degree, ends + endsLayerStride,
                                               att + (KHOP + 1), ws1);

    // Classifier + log_softmax
    gemm_bias_wmma<true, false><<<gGemmC, blk, 0, stream>>>(ws1, W2, b2, logits, NN, HID, CLS);
    log_softmax_inplace<<<gWave, blk, 0, stream>>>(logits);
}
